// PSF_44315472560245
// MI455X (gfx1250) — matrix operand order repaired and checked by reading, not compiled, not run on hardware
//
#include <hip/hip_runtime.h>
#include <hip/hip_bf16.h>

typedef __attribute__((ext_vector_type(16))) _Float16 v16h;
typedef __attribute__((ext_vector_type(8)))  float    v8f;
typedef unsigned int u32;
typedef __attribute__((ext_vector_type(4))) u32 u32x4;
typedef __attribute__((ext_vector_type(8))) int i32x8;
typedef __attribute__((ext_vector_type(4))) int i32x4;

#define PTOT 131072          // B*H*W pixels
#define HWSZ 65536           // H*W
#define CCH  128
#define TS   32
#define HALO 3
#define TL   38              // TS + 2*HALO
#define NPIX 32              // pixels per GEMM block

#if __has_builtin(__builtin_amdgcn_tensor_load_to_lds) && __has_builtin(__builtin_amdgcn_s_wait_tensorcnt)
#define USE_TDM 1
#else
#define USE_TDM 0
#endif

// ---------------------------------------------------------------------------
// Kernel 0: pack We/Wf/Wg/Wh (f32, [o][c]) into f16 A-fragments matching the
// 16x32 A-tile layout with contiguous k order (k = 16*(lane/16) + t, same as B).  Flat index:
//   idx = m*16384 + mt*2048 + ks*512 + lane*16 + t
__global__ void __launch_bounds__(256)
pack_weights_kernel(const float* __restrict__ We, const float* __restrict__ Wf,
                    const float* __restrict__ Wg, const float* __restrict__ Wh,
                    _Float16* __restrict__ packA) {
  int idx  = blockIdx.x * 256 + threadIdx.x;     // < 65536
  int t    = idx & 15;
  int lane = (idx >> 4) & 31;
  int ks   = (idx >> 9) & 3;
  int mt   = (idx >> 11) & 7;
  int m    = idx >> 14;
  const float* W = (m == 0) ? We : (m == 1) ? Wf : (m == 2) ? Wg : Wh;
  int M  = mt * 16 + (lane & 15);
  int lo = (lane & 16) ? 16 : 0;
  int Kl = lo + t;
  int K  = ks * 32 + Kl;
  packA[idx] = (_Float16)W[M * CCH + K];
}

// ---------------------------------------------------------------------------
// Kernel 1: fused 7x7 window mean/std (zero-padded SAME, n=49) + normalize
// + adain, packing f16 activations pixel-major [p][c].
__global__ void __launch_bounds__(256)
stats_kernel(const float* __restrict__ front, const float* __restrict__ back,
             _Float16* __restrict__ Xa, _Float16* __restrict__ Xf,
             _Float16* __restrict__ Xb, _Float16* __restrict__ Xh) {
  __shared__ float rf[TL][TL + 1];
  __shared__ float rb[TL][TL + 1];
  __shared__ float hf[TL][TS + 1];
  __shared__ float hf2[TL][TS + 1];
  __shared__ float hb[TL][TS + 1];
  __shared__ float hb2[TL][TS + 1];

  int t     = blockIdx.x;
  int tile  = t & 63;            // 8x8 tiles per 256x256 plane
  int plane = t >> 6;            // b*128 + c
  int c     = plane & 127;
  int b     = plane >> 7;
  int ty    = (tile >> 3) * TS;
  int tx    = (tile & 7) * TS;
  const float* fp = front + (size_t)plane * HWSZ;
  const float* bp = back  + (size_t)plane * HWSZ;
  int tid = threadIdx.x;

  for (int i = tid; i < TL * TL; i += 256) {
    int ly = i / TL, lx = i % TL;
    int gy = ty + ly - HALO, gx = tx + lx - HALO;
    bool ok = (gy >= 0) & (gy < 256) & (gx >= 0) & (gx < 256);
    rf[ly][lx] = ok ? fp[gy * 256 + gx] : 0.0f;
    rb[ly][lx] = ok ? bp[gy * 256 + gx] : 0.0f;
  }
  __syncthreads();

  for (int i = tid; i < TL * TS; i += 256) {
    int ly = i / TS, lx = i % TS;
    float sf = 0.f, sf2 = 0.f, sb = 0.f, sb2 = 0.f;
#pragma unroll
    for (int d = 0; d < 7; ++d) {
      float vf = rf[ly][lx + d], vb = rb[ly][lx + d];
      sf += vf; sf2 += vf * vf; sb += vb; sb2 += vb * vb;
    }
    hf[ly][lx] = sf; hf2[ly][lx] = sf2; hb[ly][lx] = sb; hb2[ly][lx] = sb2;
  }
  __syncthreads();

  const float inv_n = 1.0f / 49.0f;
  for (int i = tid; i < TS * TS; i += 256) {
    int y = i / TS, x = i % TS;
    float Sf = 0.f, Sf2 = 0.f, Sb = 0.f, Sb2 = 0.f;
#pragma unroll
    for (int d = 0; d < 7; ++d) {
      Sf += hf[y + d][x];  Sf2 += hf2[y + d][x];
      Sb += hb[y + d][x];  Sb2 += hb2[y + d][x];
    }
    float mf  = Sf * inv_n;
    float vfr = fmaxf(Sf2 * inv_n - mf * mf, 0.0f);
    float sdf = sqrtf(vfr + 1e-5f);
    float mb  = Sb * inv_n;
    float vbk = fmaxf(Sb2 * inv_n - mb * mb, 0.0f);
    float sdb = sqrtf(vbk + 1e-5f);
    float fv  = rf[y + HALO][x + HALO];
    float bv  = rb[y + HALO][x + HALO];
    float fn  = (fv - mf) / sdf;
    float bn  = (bv - mb) / sdb;
    float ad  = sdb * fn + mb;
    size_t p   = (size_t)b * HWSZ + (size_t)(ty + y) * 256 + (tx + x);
    size_t ix  = p * CCH + c;
    Xa[ix] = (_Float16)ad;
    Xf[ix] = (_Float16)fn;
    Xb[ix] = (_Float16)bn;
    Xh[ix] = (_Float16)bv;
  }
}

// ---------------------------------------------------------------------------
// Helper: issue one TDM 1-D tile load (contiguous `nelem` f16) into LDS.
#if USE_TDM
__device__ __forceinline__ void tdm_load_1d(const _Float16* gsrc, void* lds_dst,
                                            unsigned nelem) {
  unsigned lds_off = (unsigned)(unsigned long long)lds_dst;  // low 32b = LDS byte addr
  unsigned long long ga = (unsigned long long)gsrc;
  u32x4 g0;
  g0.x = 1u;                                   // count=1, user descriptor
  g0.y = lds_off;                              // lds_addr (bytes)
  g0.z = (u32)ga;                              // global_addr[31:0]
  g0.w = (u32)((ga >> 32) & 0x01FFFFFFu) | (2u << 30);  // addr[56:32] | type=2
  i32x8 g1 = {0, 0, 0, 0, 0, 0, 0, 0};
  g1[0] = (int)(1u << 16);                     // data_size=1 (2 bytes/elem)
  g1[1] = (int)((nelem & 0xFFFFu) << 16);      // tensor_dim0[15:0]
  g1[2] = (int)((nelem >> 16) & 0xFFFFu) | (1 << 16);  // tensor_dim0[31:16], tensor_dim1=1
  g1[3] = (int)((nelem & 0xFFFFu) << 16);      // tile_dim0 (1-D tile)
  g1[5] = (int)nelem;                          // tensor_dim0_stride
  i32x4 z4 = {0, 0, 0, 0};
  i32x8 z8 = {0, 0, 0, 0, 0, 0, 0, 0};
  __builtin_amdgcn_tensor_load_to_lds(g0, g1, z4, z4, z8, 0);
}
#endif

// ---------------------------------------------------------------------------
// Kernel 2: fused quad-GEMM per 32-pixel tile.  8 waves = 8 M-tiles (128 out
// channels).  Each wave: 4 GEMMs x 2 N-subtiles x 4 K-steps = 32 WMMA f16->f32.
// B operands staged to LDS by the Tensor Data Mover (TENSORcnt-tracked).
// FF/GG reduced in-block to cosine-similarity S; EE/HH written f32 [o][p].
__global__ void __launch_bounds__(256)
gemm_fused_kernel(const _Float16* __restrict__ Xa, const _Float16* __restrict__ Xf,
                  const _Float16* __restrict__ Xb, const _Float16* __restrict__ Xh,
                  const _Float16* __restrict__ packA,
                  const float* __restrict__ be, const float* __restrict__ bf,
                  const float* __restrict__ bg, const float* __restrict__ bh,
                  float* __restrict__ EE, float* __restrict__ HH,
                  float* __restrict__ Sout) {
  __shared__ __align__(32) _Float16 sB[4 * NPIX * CCH];   // 32 KB
  __shared__ float sDot[NPIX], sF2[NPIX], sG2[NPIX];

  const int p0  = blockIdx.x * NPIX;
  const int tid = threadIdx.x;

#if USE_TDM
  // Wave 0 issues one TDM descriptor per source buffer (8 KB contiguous each).
  if (tid < 32) {
    tdm_load_1d(Xa + (size_t)p0 * CCH, sB + 0 * NPIX * CCH, NPIX * CCH);
    tdm_load_1d(Xf + (size_t)p0 * CCH, sB + 1 * NPIX * CCH, NPIX * CCH);
    tdm_load_1d(Xb + (size_t)p0 * CCH, sB + 2 * NPIX * CCH, NPIX * CCH);
    tdm_load_1d(Xh + (size_t)p0 * CCH, sB + 3 * NPIX * CCH, NPIX * CCH);
    __builtin_amdgcn_s_wait_tensorcnt(0);
  }
#else
#pragma unroll
  for (int g = 0; g < 4; ++g) {
    const _Float16* src16 = (g == 0) ? Xa : (g == 1) ? Xf : (g == 2) ? Xb : Xh;
    const unsigned int* src = (const unsigned int*)src16;
    unsigned int* dst = (unsigned int*)(sB + g * NPIX * CCH);
    for (int i = tid; i < NPIX * CCH / 2; i += 256)
      dst[i] = src[(size_t)p0 * 64 + i];
  }
#endif
  if (tid < NPIX) { sDot[tid] = 0.f; sF2[tid] = 0.f; sG2[tid] = 0.f; }
  __syncthreads();

  const int lane = tid & 31;
  const int mt   = tid >> 5;
  const int nloc = lane & 15;
  const int hiK  = (lane & 16) ? 16 : 0;

  v8f aE[2] = {{}, {}}, aF[2] = {{}, {}}, aG[2] = {{}, {}}, aH[2] = {{}, {}};
#pragma unroll
  for (int ks = 0; ks < 4; ++ks) {
    // A fragments (pre-swizzled, one contiguous 32B load per lane per matrix),
    // reused across both N-subtiles.
    v16h wE = *(const v16h*)(packA + ((((0 * 8 + mt) * 4 + ks) << 9) + lane * 16));
    v16h wF = *(const v16h*)(packA + ((((1 * 8 + mt) * 4 + ks) << 9) + lane * 16));
    v16h wG = *(const v16h*)(packA + ((((2 * 8 + mt) * 4 + ks) << 9) + lane * 16));
    v16h wH = *(const v16h*)(packA + ((((3 * 8 + mt) * 4 + ks) << 9) + lane * 16));
#pragma unroll
    for (int ns = 0; ns < 2; ++ns) {
      int boff = (ns * 16 + nloc) * CCH + ks * 32 + hiK;
      v16h b0 = *(const v16h*)(sB + 0 * NPIX * CCH + boff);
      v16h b1 = *(const v16h*)(sB + 1 * NPIX * CCH + boff);
      v16h b2 = *(const v16h*)(sB + 2 * NPIX * CCH + boff);
      v16h b3 = *(const v16h*)(sB + 3 * NPIX * CCH + boff);
      aE[ns] = __builtin_amdgcn_wmma_f32_16x16x32_f16(false, wE, false, b0, (short)0, aE[ns], false, false);
      aF[ns] = __builtin_amdgcn_wmma_f32_16x16x32_f16(false, wF, false, b1, (short)0, aF[ns], false, false);
      aG[ns] = __builtin_amdgcn_wmma_f32_16x16x32_f16(false, wG, false, b2, (short)0, aG[ns], false, false);
      aH[ns] = __builtin_amdgcn_wmma_f32_16x16x32_f16(false, wH, false, b3, (short)0, aH[ns], false, false);
    }
  }

  // C/D layout: VGPR j <-> M = j + (lane>=16 ? 8 : 0), N = lane%16.
  const int rbase = mt * 16 + ((lane & 16) ? 8 : 0);
#pragma unroll
  for (int ns = 0; ns < 2; ++ns) {
    const int nl = ns * 16 + nloc;
    const int p  = p0 + nl;
    float dot = 0.f, f2 = 0.f, g2 = 0.f;
#pragma unroll
    for (int j = 0; j < 8; ++j) {
      int o = rbase + j;
      float e = aE[ns][j] + be[o];
      float h = aH[ns][j] + bh[o];
      EE[(size_t)o * PTOT + p] = e;
      HH[(size_t)o * PTOT + p] = h;
      float f = aF[ns][j] + bf[o];
      float g = aG[ns][j] + bg[o];
      dot += f * g; f2 += f * f; g2 += g * g;
    }
    atomicAdd(&sDot[nl], dot);
    atomicAdd(&sF2[nl], f2);
    atomicAdd(&sG2[nl], g2);
  }
  __syncthreads();
  if (tid < NPIX)
    Sout[p0 + tid] = sDot[tid] / (sqrtf(sF2[tid]) * sqrtf(sG2[tid]));
}

// ---------------------------------------------------------------------------
// Kernel 3: per-batch min/max of S.
__global__ void __launch_bounds__(256)
minmax_kernel(const float* __restrict__ S, float* __restrict__ mm) {
  __shared__ float smn[256], smx[256];
  int b = blockIdx.x, tid = threadIdx.x;
  const float* s = S + (size_t)b * HWSZ;
  float mn = 3.4e38f, mx = -3.4e38f;
  for (int i = tid; i < HWSZ; i += 256) {
    float v = s[i];
    mn = fminf(mn, v); mx = fmaxf(mx, v);
  }
  smn[tid] = mn; smx[tid] = mx;
  __syncthreads();
  for (int off = 128; off > 0; off >>= 1) {
    if (tid < off) {
      smn[tid] = fminf(smn[tid], smn[tid + off]);
      smx[tid] = fmaxf(smx[tid], smx[tid + off]);
    }
    __syncthreads();
  }
  if (tid == 0) { mm[b * 2] = smn[0]; mm[b * 2 + 1] = smx[0]; }
}

// ---------------------------------------------------------------------------
// Kernel 4: O[:, :128] = S_n*EE + (1-S_n)*HH ; O[:, 128] = mask.
__global__ void __launch_bounds__(256)
combine_kernel(const float* __restrict__ EE, const float* __restrict__ HH,
               const float* __restrict__ S, const float* __restrict__ mm,
               const float* __restrict__ mask, float* __restrict__ out) {
  size_t idx = (size_t)blockIdx.x * 256 + threadIdx.x;
  const size_t per_b = (size_t)129 * HWSZ;
  int b  = (int)(idx / per_b);
  size_t r = idx % per_b;
  int o  = (int)(r / HWSZ);
  int hw = (int)(r % HWSZ);
  size_t p = (size_t)b * HWSZ + hw;
  if (o == 128) { out[idx] = mask[p]; return; }
  float smin = mm[b * 2], smax = mm[b * 2 + 1];
  float sn = (S[p] - smin) / (smax - smin);
  out[idx] = sn * EE[(size_t)o * PTOT + p] + (1.0f - sn) * HH[(size_t)o * PTOT + p];
}

// ---------------------------------------------------------------------------
extern "C" void kernel_launch(void* const* d_in, const int* in_sizes, int n_in,
                              void* d_out, int out_size, void* d_ws, size_t ws_size,
                              hipStream_t stream) {
  const float* front = (const float*)d_in[0];
  const float* back  = (const float*)d_in[1];
  const float* mask  = (const float*)d_in[2];
  const float* We    = (const float*)d_in[3];
  const float* be    = (const float*)d_in[4];
  const float* Wf    = (const float*)d_in[5];
  const float* bf    = (const float*)d_in[6];
  const float* Wg    = (const float*)d_in[7];
  const float* bg    = (const float*)d_in[8];
  const float* Wh    = (const float*)d_in[9];
  const float* bh    = (const float*)d_in[10];
  float* out = (float*)d_out;

  char* ws = (char*)d_ws;
  const size_t Xelem = (size_t)PTOT * CCH;          // 16,777,216 halves each
  _Float16* Xa = (_Float16*)ws;
  _Float16* Xf = Xa + Xelem;
  _Float16* Xb = Xf + Xelem;
  _Float16* Xh = Xb + Xelem;
  float* EE = (float*)(Xh + Xelem);                 // offset 128 MB
  float* HH = EE + Xelem;                           // offset 192 MB
  float* S  = HH + Xelem;                           // offset 256 MB
  _Float16* packA = (_Float16*)(S + PTOT);
  float* mm = (float*)(packA + 65536);

  pack_weights_kernel<<<256, 256, 0, stream>>>(We, Wf, Wg, Wh, packA);
  stats_kernel<<<2 * CCH * 64, 256, 0, stream>>>(front, back, Xa, Xf, Xb, Xh);
  gemm_fused_kernel<<<PTOT / NPIX, 256, 0, stream>>>(Xa, Xf, Xb, Xh, packA,
                                                     be, bf, bg, bh, EE, HH, S);
  minmax_kernel<<<2, 256, 0, stream>>>(S, mm);
  const size_t total = (size_t)2 * 129 * HWSZ;      // 16,908,288
  combine_kernel<<<(unsigned)((total + 255) / 256), 256, 0, stream>>>(EE, HH, S, mm, mask, out);
}
